// FullAttention_34634616275457
// MI455X (gfx1250) — matrix operand order repaired and checked by reading, not compiled, not run on hardware
//
#include <hip/hip_runtime.h>

#define B_ 2
#define L_ 2048
#define H_ 16
#define E_ 64
#define QT_ (L_ / 16)          // 128 query tiles per (b,h)
#define SCALE 0.125f           // 1/sqrt(64)
#define NEG_BIG (-1.0e30f)

typedef __attribute__((ext_vector_type(16))) _Float16 v16h;
typedef __attribute__((ext_vector_type(8)))  float    v8f;
typedef __attribute__((ext_vector_type(4)))  float    v4f;

// K-index of element i (0..15) of an f16 A-fragment, contiguous order:
// lanes 0-15: K = 0..15; lanes 16-31: K = 16..31
__device__ __forceinline__ int a_k(int i, int half) {
    return i + 16 * half;
}

__device__ __forceinline__ v8f wmma_f16(v16h a, v16h b, v8f c) {
    return __builtin_amdgcn_wmma_f32_16x16x32_f16(false, a, false, b, (short)0, c,
                                                  false, false);
}

__global__ __launch_bounds__(128) void flash_attn_fwd(
    const float* __restrict__ Q, const float* __restrict__ Kk,
    const float* __restrict__ Vv, float* __restrict__ Vout,
    float* __restrict__ Aout)
{
    __shared__ _Float16 pbuf[4][16][32];   // per-wave P tile (16 rows x 32 keys)

    const int lane = threadIdx.x & 31;
    const int wid  = threadIdx.x >> 5;
    const int half = lane >> 4;
    const int m16  = lane & 15;

    const int tile   = blockIdx.x * 4 + wid;       // 0..4095
    const int q_tile = tile & (QT_ - 1);
    const int h      = (tile >> 7) & (H_ - 1);
    const int b      = tile >> 11;
    const int qbase  = q_tile * 16;
    const int qlast  = qbase + 15;

    // ---------------- Q A-fragments (f32 -> f16), E split in two K=32 chunks
    const float* qrow = Q + (((size_t)b * L_ + (qbase + m16)) * H_ + h) * E_;
    v16h qa0, qa1;
    #pragma unroll
    for (int i = 0; i < 16; ++i) {
        int k = a_k(i, half);
        qa0[i] = (_Float16)qrow[k];
        qa1[i] = (_Float16)qrow[k + 32];
    }

    // ---------------- Pass A: per-lane online softmax partials (NO cross-lane
    // traffic in the loop; each lane owns one column per tile)
    float mrow[8], lrow[8];
    #pragma unroll
    for (int j = 0; j < 8; ++j) { mrow[j] = NEG_BIG; lrow[j] = 0.0f; }

    const int ntiles = q_tile + 1;
    for (int st = 0; st < ntiles; ++st) {
        const int s0 = st * 16;
        const float* krow = Kk + (((size_t)b * L_ + (s0 + m16)) * H_ + h) * E_;
        v16h kb0, kb1;
        #pragma unroll
        for (int i = 0; i < 16; ++i) {
            kb0[i] = (_Float16)krow[i + 16 * half];
            kb1[i] = (_Float16)krow[i + 16 * half + 32];
        }
        v8f c = {};
        c = wmma_f16(qa0, kb0, c);
        c = wmma_f16(qa1, kb1, c);

        const int col = s0 + m16;
        #pragma unroll
        for (int j = 0; j < 8; ++j) {
            const int row = qbase + j + 8 * half;
            const bool masked = (col > row);
            const float s  = SCALE * c[j];
            const float mo = mrow[j];
            const float mn = masked ? mo : fmaxf(mo, s);
            // mo <= mn always; if both are NEG_BIG, corr==1 but lrow==0 -> safe
            lrow[j] = lrow[j] * __expf(mo - mn) +
                      (masked ? 0.0f : __expf(s - mn));
            mrow[j] = mn;
        }
    }

    // ---------------- single cross-lane (m, l) merge: 4 butterfly steps
    #pragma unroll
    for (int j = 0; j < 8; ++j) {
        #pragma unroll
        for (int d = 1; d < 16; d <<= 1) {
            const float mo2 = __shfl_xor(mrow[j], d, 32);
            const float lo2 = __shfl_xor(lrow[j], d, 32);
            const float mn  = fmaxf(mrow[j], mo2);
            // lanes with (NEG_BIG, 0) partials contribute exactly 0
            lrow[j] = lrow[j] * __expf(mrow[j] - mn) + lo2 * __expf(mo2 - mn);
            mrow[j] = mn;
        }
    }

    float rinv[8];
    #pragma unroll
    for (int j = 0; j < 8; ++j) rinv[j] = 1.0f / lrow[j];

    // ---------------- Pass B: recompute S, write normalized P to A, O += P*V
    float* abase = Aout + ((size_t)(b * H_ + h) * L_ + qbase) * L_;
    v8f o[4] = {v8f{}, v8f{}, v8f{}, v8f{}};

    const int nblk = qlast / 32 + 1;   // 32-key blocks containing any unmasked work
    for (int blk = 0; blk < nblk; ++blk) {
        const int sb = blk * 32;
        #pragma unroll
        for (int sub = 0; sub < 2; ++sub) {
            const int s0 = sb + sub * 16;
            if (s0 <= qlast) {
                const float* krow =
                    Kk + (((size_t)b * L_ + (s0 + m16)) * H_ + h) * E_;
                v16h kb0, kb1;
                #pragma unroll
                for (int i = 0; i < 16; ++i) {
                    kb0[i] = (_Float16)krow[i + 16 * half];
                    kb1[i] = (_Float16)krow[i + 16 * half + 32];
                }
                v8f c = {};
                c = wmma_f16(qa0, kb0, c);
                c = wmma_f16(qa1, kb1, c);

                const int col = s0 + m16;
                #pragma unroll
                for (int j = 0; j < 8; ++j) {
                    const int row = j + 8 * half;
                    float p = (col > qbase + row)
                                  ? 0.0f
                                  : __expf(SCALE * c[j] - mrow[j]) * rinv[j];
                    // A is write-once streaming output: don't pollute L2
                    __builtin_nontemporal_store(p, &abase[(size_t)row * L_ + col]);
                    pbuf[wid][row][sub * 16 + m16] = (_Float16)p;
                }
            } else {
                const int col = s0 + m16;
                #pragma unroll
                for (int j = 0; j < 8; ++j) {
                    const int row = j + 8 * half;
                    __builtin_nontemporal_store(0.0f,
                                                &abase[(size_t)row * L_ + col]);
                    pbuf[wid][row][sub * 16 + m16] = (_Float16)0.0f;
                }
            }
        }
        asm volatile("" ::: "memory");   // keep LDS stores before the transposed reads

        // P in A-fragment layout (LDS round-trip = C-layout -> A-layout transpose)
        v16h pa;
        #pragma unroll
        for (int i = 0; i < 16; ++i)
            pa[i] = pbuf[wid][m16][a_k(i, half)];

        // O(16x64) += P(16x32) * V(32x64), four N=16 tiles of V
        #pragma unroll
        for (int t = 0; t < 4; ++t) {
            v16h vb;
            #pragma unroll
            for (int i = 0; i < 16; ++i) {
                const int s = sb + i + 16 * half;   // always < L_
                vb[i] = (_Float16)
                    Vv[(((size_t)b * L_ + s) * H_ + h) * E_ + t * 16 + m16];
            }
            o[t] = wmma_f16(pa, vb, o[t]);
        }
    }

    // ---------------- zero-fill the fully masked region of A (streaming NT)
    const int zstart = nblk * 32;
    for (int r = 0; r < 16; ++r) {
        float* arow = abase + (size_t)r * L_;
        for (int cidx = zstart + lane * 4; cidx < L_; cidx += 128) {
            const v4f z = {0.0f, 0.0f, 0.0f, 0.0f};
            __builtin_nontemporal_store(z, (v4f*)(arow + cidx));
        }
    }

    // ---------------- write V output: O / l
    #pragma unroll
    for (int t = 0; t < 4; ++t) {
        #pragma unroll
        for (int j = 0; j < 8; ++j) {
            const int q = qbase + j + 8 * half;
            Vout[(((size_t)b * L_ + q) * H_ + h) * E_ + t * 16 + m16] =
                o[t][j] * rinv[j];
        }
    }
}

extern "C" void kernel_launch(void* const* d_in, const int* in_sizes, int n_in,
                              void* d_out, int out_size, void* d_ws, size_t ws_size,
                              hipStream_t stream) {
    const float* Q  = (const float*)d_in[0];
    const float* Kk = (const float*)d_in[1];
    const float* Vv = (const float*)d_in[2];
    float* Vout = (float*)d_out;
    float* Aout = (float*)d_out + (size_t)B_ * L_ * H_ * E_;

    dim3 grid(1024);   // 4096 query tiles / 4 waves per block
    dim3 block(128);
    hipLaunchKernelGGL(flash_attn_fwd, grid, block, 0, stream, Q, Kk, Vv, Vout, Aout);
}
